// SpatialAttention_21543555957158
// MI455X (gfx1250) — matrix operand order repaired and checked by reading, not compiled, not run on hardware
//
#include <hip/hip_runtime.h>

// ---------------------------------------------------------------------------
// SpatialAttention for MI455X (gfx1250): bf16 WMMA everywhere,
// TDM (tensor_load_to_lds) for 2D tile staging, async-LDS fallback.
//   N=128 images, S=256 tokens, C=1024, 16 heads x 64 dim, scale=1/8.
// Workspace (bf16): xb[32768*1024] wqkvb[1024*3072] wprojb[1024*1024]
//                   qkvb[32768*3072] aob[32768*1024]   (~344 MB total)
// ---------------------------------------------------------------------------

typedef __bf16 bf16;
typedef __attribute__((ext_vector_type(16))) __bf16 v16bf;
typedef __attribute__((ext_vector_type(8)))  float  v8f;
typedef int v4i __attribute__((vector_size(4 * sizeof(int))));

#define WMMA_BF16(A, B, C) \
  __builtin_amdgcn_wmma_f32_16x16x32_bf16(false, (A), false, (B), (short)0, (C), false, false)

#define NTOK  32768
#define CDIM  1024
#define C3    3072
#define NHEAD 16
#define HDIM  64

typedef __attribute__((address_space(1))) v4i* gptr_v4i;
typedef __attribute__((address_space(3))) v4i* lptr_v4i;

// ---- async global->LDS copy (GLOBAL_LOAD_ASYNC_TO_LDS_B128), fallback path --
#if __has_builtin(__builtin_amdgcn_global_load_async_to_lds_b128) && \
    __has_builtin(__builtin_amdgcn_s_wait_asynccnt)
#define HAVE_ASYNC_LDS 1
__device__ inline void async_copy16(const bf16* g, bf16* l) {
  __builtin_amdgcn_global_load_async_to_lds_b128((gptr_v4i)(g), (lptr_v4i)(l), 0, 0);
}
__device__ inline void async_wait() { __builtin_amdgcn_s_wait_asynccnt(0); }
#else
__device__ inline void async_copy16(const bf16* g, bf16* l) {
  *(uint4*)l = *(const uint4*)g;
}
__device__ inline void async_wait() {}
#endif

// ---- Tensor Data Mover: one-instruction 2D tile load into LDS ---------------
#if __has_builtin(__builtin_amdgcn_tensor_load_to_lds) && \
    __has_builtin(__builtin_amdgcn_s_wait_tensorcnt)
#define HAVE_TDM 1
typedef unsigned int v4u __attribute__((vector_size(16)));
typedef int v8i __attribute__((vector_size(32)));

__device__ inline unsigned lds_byte_off(const void* p) {
  return (unsigned)(unsigned long long)
      (__attribute__((address_space(3))) const void*)p;
}

// 2D tile: tile_w bf16 elements per row, tile_h rows, row stride in elements.
// Tile is fully in-bounds, so tensor dims == tile dims.
__device__ inline void tdm_load_2d(unsigned lds_off, const void* gptr,
                                   unsigned tile_w, unsigned tile_h,
                                   unsigned long long stride_elems) {
  unsigned long long ga = (unsigned long long)gptr;
  v4u g0;
  g0[0] = 1u;                                               // count=1 (valid D#)
  g0[1] = lds_off;                                          // lds_addr (bytes)
  g0[2] = (unsigned)ga;                                     // global_addr[31:0]
  g0[3] = (unsigned)((ga >> 32) & 0x01FFFFFFu) | (2u << 30); // addr[56:32]|type=2
  // group1 bits[255:0] assembled as 4x u64
  unsigned long long q0 =
      (1ull << 16)                                          // data_size=1 -> 2B
    | ((unsigned long long)(tile_w & 0xFFFFu) << 48);       // tensor_dim0[15:0]
  unsigned long long q1 =
      ((unsigned long long)(tile_w >> 16) & 0xFFFFull)      // tensor_dim0[31:16]
    | ((unsigned long long)tile_h << 16)                    // tensor_dim1[31:0]
    | ((unsigned long long)(tile_w & 0xFFFFu) << 48);       // tile_dim0
  unsigned long long q2 =
      (unsigned long long)(tile_h & 0xFFFFu)                // tile_dim1
    | ((stride_elems & 0xFFFFFFFFull) << 32);               // dim0_stride[31:0]
  unsigned long long q3 = (stride_elems >> 32) & 0xFFFFull; // dim0_stride[47:32]
  v8i g1;
  g1[0] = (int)(unsigned)q0; g1[1] = (int)(unsigned)(q0 >> 32);
  g1[2] = (int)(unsigned)q1; g1[3] = (int)(unsigned)(q1 >> 32);
  g1[4] = (int)(unsigned)q2; g1[5] = (int)(unsigned)(q2 >> 32);
  g1[6] = (int)(unsigned)q3; g1[7] = (int)(unsigned)(q3 >> 32);
  v4i z = {0, 0, 0, 0};
  v8i z8 = {0, 0, 0, 0, 0, 0, 0, 0};
  __builtin_amdgcn_tensor_load_to_lds(g0, g1, z, z, z8, 0);
}
__device__ inline void tdm_wait() { __builtin_amdgcn_s_wait_tensorcnt(0); }
#endif

// ---- fragment loaders (LDS -> VGPR), per CDNA5 16-bit WMMA layouts ----------
__device__ inline v16bf load_a_frag(const bf16* p, int half) {
  union { uint4 u[2]; v16bf v; } t;
  t.u[0] = *(const uint4*)(p + 8 * half);
  t.u[1] = *(const uint4*)(p + 16 + 8 * half);
  return t.v;
}
__device__ inline v16bf load_b_frag(const bf16* p, int half) {
  union { uint4 u[2]; v16bf v; } t;
  t.u[0] = *(const uint4*)(p + 8 * half);
  t.u[1] = *(const uint4*)(p + 16 + 8 * half);
  return t.v;
}

// ---- fp32 -> bf16 convert ---------------------------------------------------
__global__ void cvt_f32_bf16(const float4* __restrict__ in,
                             unsigned long long* __restrict__ out, int n4) {
  int i = blockIdx.x * blockDim.x + threadIdx.x;
  if (i >= n4) return;
  float4 f = in[i];
  union { bf16 h[4]; unsigned long long u; } u;
  u.h[0] = (bf16)f.x; u.h[1] = (bf16)f.y; u.h[2] = (bf16)f.z; u.h[3] = (bf16)f.w;
  out[i] = u.u;
}

// ---- tiled WMMA GEMM: C[MxN] = A[MxK] * B[KxN] ------------------------------
// BM=128 BN=64 BK=64, 8 waves (4xM, 2xN), each wave 32x32; 8 WMMAs per K-step.
template <bool BF16OUT>
__launch_bounds__(256)
__global__ void gemm_wmma(const bf16* __restrict__ A, const bf16* __restrict__ B,
                          void* __restrict__ Cvoid, const float* __restrict__ bias,
                          int M, int N, int K) {
  __shared__ bf16 ldsA[128 * 64];   // 16 KB
  __shared__ bf16 ldsBt[64 * 64];   //  8 KB, B transposed: [n][k]
  const int tid  = threadIdx.x;
  const int wave = tid >> 5, lane = tid & 31;
  const int half = lane >> 4, lr = lane & 15;
  const int wm = wave & 3, wn = wave >> 2;
  const int m0 = blockIdx.y * 128, n0 = blockIdx.x * 64;

  v8f acc[2][2] = {};
  for (int k0 = 0; k0 < K; k0 += 64) {
    // A tile 128x64: one TDM descriptor (wave 0), else per-thread async
#ifdef HAVE_TDM
    if (wave == 0)
      tdm_load_2d(lds_byte_off(ldsA), &A[(size_t)m0 * K + k0], 64, 128,
                  (unsigned long long)K);
#else
#pragma unroll
    for (int it = 0; it < 4; ++it) {
      int c = tid + 256 * it;
      int row = c >> 3, kc = c & 7;
      async_copy16(&A[(size_t)(m0 + row) * K + k0 + kc * 8],
                   &ldsA[row * 64 + kc * 8]);
    }
#endif
    // B tile 64x64 transposed: vector read, LDS scatter (all threads)
#pragma unroll
    for (int it = 0; it < 2; ++it) {
      int c = tid + 256 * it;
      int k = c >> 3, nc = c & 7;
      union { uint4 u; bf16 h[8]; } t;
      t.u = *(const uint4*)&B[(size_t)(k0 + k) * N + n0 + nc * 8];
#pragma unroll
      for (int i = 0; i < 8; ++i) ldsBt[(nc * 8 + i) * 64 + k] = t.h[i];
    }
#ifdef HAVE_TDM
    tdm_wait();
#else
    async_wait();
#endif
    __syncthreads();
    v16bf af[2][2], bfr[2][2];
#pragma unroll
    for (int i = 0; i < 2; ++i)
#pragma unroll
      for (int kk = 0; kk < 2; ++kk)
        af[i][kk] = load_a_frag(&ldsA[(wm * 32 + i * 16 + lr) * 64 + kk * 32], half);
#pragma unroll
    for (int j = 0; j < 2; ++j)
#pragma unroll
      for (int kk = 0; kk < 2; ++kk)
        bfr[j][kk] = load_b_frag(&ldsBt[(wn * 32 + j * 16 + lr) * 64 + kk * 32], half);
#pragma unroll
    for (int kk = 0; kk < 2; ++kk)
#pragma unroll
      for (int i = 0; i < 2; ++i)
#pragma unroll
        for (int j = 0; j < 2; ++j)
          acc[i][j] = WMMA_BF16(af[i][kk], bfr[j][kk], acc[i][j]);
    __syncthreads();
  }
  // epilogue
#pragma unroll
  for (int i = 0; i < 2; ++i)
#pragma unroll
    for (int j = 0; j < 2; ++j) {
      int row0 = m0 + wm * 32 + i * 16 + 8 * half;
      int col  = n0 + wn * 32 + j * 16 + lr;
      size_t base = (size_t)row0 * N + col;
      if constexpr (BF16OUT) {
        bf16* C = (bf16*)Cvoid;
#pragma unroll
        for (int r = 0; r < 8; ++r) C[base + (size_t)r * N] = (bf16)acc[i][j][r];
      } else {
        float* C = (float*)Cvoid;
        float bv = bias[col];
#pragma unroll
        for (int r = 0; r < 8; ++r) C[base + (size_t)r * N] = acc[i][j][r] + bv;
      }
    }
}

// ---- in-place 2D RoPE on q,k slices of qkv (bf16) ---------------------------
__global__ void rope_kernel(bf16* __restrict__ qkv) {
  unsigned gid = blockIdx.x * 256u + threadIdx.x;  // 33,554,432 total
  int j    = gid & 15;  gid >>= 4;
  int part = gid & 1;   gid >>= 1;
  int qk   = gid & 1;   gid >>= 1;
  int h    = gid & 15;  gid >>= 4;
  int tok  = gid;
  int s = tok & 255;
  int pos = part ? (s & 15) : (s >> 4);
  float f = __powf(10000.0f, -(float)j * (1.0f / 16.0f));
  float sn, cs;
  __sincosf((float)pos * f, &sn, &cs);
  size_t base = (size_t)tok * C3 + (size_t)qk * CDIM + h * HDIM + part * 32 + j;
  float x1 = (float)qkv[base], x2 = (float)qkv[base + 16];
  qkv[base]      = (bf16)(x1 * cs - x2 * sn);
  qkv[base + 16] = (bf16)(x2 * cs + x1 * sn);
}

// ---- flash attention: one block per (image, head, 128-query-row half) -------
__launch_bounds__(256)
__global__ void attn_wmma(const bf16* __restrict__ qkv, bf16* __restrict__ ao) {
  __shared__ bf16 ldsQ[128 * 64];    // 16 KB
  __shared__ bf16 ldsK[128 * 64];    // 16 KB
  __shared__ bf16 ldsVT[64 * 128];   // 16 KB transposed V: [d][t]
  __shared__ bf16 ldsP[8 * 16 * 32]; //  8 KB per-wave prob scratch
  const int tid  = threadIdx.x;
  const int wave = tid >> 5, lane = tid & 31;
  const int half = lane >> 4, lr = lane & 15;
  const int m0 = wave * 16;
  const int bid = blockIdx.x;
  const int n = bid >> 5, h = (bid >> 1) & 15, qhalf = bid & 1;

  const bf16* qp = qkv + (size_t)(n * 256 + qhalf * 128) * C3 + h * HDIM;
  const bf16* kp = qkv + (size_t)(n * 256) * C3 + CDIM + h * HDIM;
  const bf16* vp = kp + CDIM;

  // stage Q (128 x 64)
#ifdef HAVE_TDM
  if (wave == 0) tdm_load_2d(lds_byte_off(ldsQ), qp, 64, 128, C3);
  tdm_wait();
#else
#pragma unroll
  for (int it = 0; it < 4; ++it) {
    int c = tid + 256 * it;
    int t = c >> 3, dc = c & 7;
    async_copy16(&qp[(size_t)t * C3 + dc * 8], &ldsQ[t * 64 + dc * 8]);
  }
  async_wait();
#endif
  __syncthreads();

  v16bf qf[2];
#pragma unroll
  for (int j = 0; j < 2; ++j)
    qf[j] = load_a_frag(&ldsQ[(m0 + lr) * 64 + j * 32], half);

  float m_i[8], l_i[8];
  v8f o[4] = {};
#pragma unroll
  for (int r = 0; r < 8; ++r) { m_i[r] = -1e30f; l_i[r] = 0.0f; }

  for (int stg = 0; stg < 2; ++stg) {
    __syncthreads();
    // stage K (TDM or async) and transposed V (manual scatter, all threads)
#ifdef HAVE_TDM
    if (wave == 0)
      tdm_load_2d(lds_byte_off(ldsK), &kp[(size_t)(stg * 128) * C3], 64, 128, C3);
#endif
#pragma unroll
    for (int it = 0; it < 4; ++it) {
      int c = tid + 256 * it;
      int t = c >> 3, dc = c & 7;
#ifndef HAVE_TDM
      async_copy16(&kp[(size_t)(stg * 128 + t) * C3 + dc * 8],
                   &ldsK[t * 64 + dc * 8]);
#endif
      union { uint4 u; bf16 hh[8]; } v;
      v.u = *(const uint4*)&vp[(size_t)(stg * 128 + t) * C3 + dc * 8];
#pragma unroll
      for (int i = 0; i < 8; ++i) ldsVT[(dc * 8 + i) * 128 + t] = v.hh[i];
    }
#ifdef HAVE_TDM
    tdm_wait();
#else
    async_wait();
#endif
    __syncthreads();

    for (int kb = 0; kb < 4; ++kb) {
      const int t0 = kb * 32;
      v8f c0 = {}, c1 = {};
#pragma unroll
      for (int j = 0; j < 2; ++j) {
        v16bf kf0 = load_b_frag(&ldsK[(t0 + lr) * 64 + j * 32], half);
        v16bf kf1 = load_b_frag(&ldsK[(t0 + 16 + lr) * 64 + j * 32], half);
        c0 = WMMA_BF16(qf[j], kf0, c0);
        c1 = WMMA_BF16(qf[j], kf1, c1);
      }
      // online softmax
#pragma unroll
      for (int r = 0; r < 8; ++r) {
        float s0 = c0[r] * 0.125f, s1 = c1[r] * 0.125f;
        float mx = fmaxf(s0, s1);
        mx = fmaxf(mx, __shfl_xor(mx, 1, 32));
        mx = fmaxf(mx, __shfl_xor(mx, 2, 32));
        mx = fmaxf(mx, __shfl_xor(mx, 4, 32));
        mx = fmaxf(mx, __shfl_xor(mx, 8, 32));
        float mnew  = fmaxf(m_i[r], mx);
        float alpha = __expf(m_i[r] - mnew);
        float p0 = __expf(s0 - mnew), p1 = __expf(s1 - mnew);
        float rs = p0 + p1;
        rs += __shfl_xor(rs, 1, 32);
        rs += __shfl_xor(rs, 2, 32);
        rs += __shfl_xor(rs, 4, 32);
        rs += __shfl_xor(rs, 8, 32);
        l_i[r] = l_i[r] * alpha + rs;
        m_i[r] = mnew;
#pragma unroll
        for (int j = 0; j < 4; ++j) o[j][r] *= alpha;
        ldsP[wave * 512 + (r + 8 * half) * 32 + lr]      = (bf16)p0;
        ldsP[wave * 512 + (r + 8 * half) * 32 + 16 + lr] = (bf16)p1;
      }
      // P(16x32) @ V(32x64)
      v16bf pf = load_a_frag(&ldsP[wave * 512 + lr * 32], half);
#pragma unroll
      for (int j = 0; j < 4; ++j) {
        v16bf vf = load_b_frag(&ldsVT[(j * 16 + lr) * 128 + t0], half);
        o[j] = WMMA_BF16(pf, vf, o[j]);
      }
    }
  }
  // normalize + write (token, head, d) bf16
#pragma unroll
  for (int r = 0; r < 8; ++r) {
    float inv = 1.0f / l_i[r];
    int tok = n * 256 + qhalf * 128 + m0 + r + 8 * half;
#pragma unroll
    for (int j = 0; j < 4; ++j)
      ao[(size_t)tok * CDIM + h * HDIM + j * 16 + lr] = (bf16)(o[j][r] * inv);
  }
}

// ---------------------------------------------------------------------------
extern "C" void kernel_launch(void* const* d_in, const int* in_sizes, int n_in,
                              void* d_out, int out_size, void* d_ws, size_t ws_size,
                              hipStream_t stream) {
  const float* x      = (const float*)d_in[0];
  const float* w_qkv  = (const float*)d_in[1];
  const float* w_proj = (const float*)d_in[2];
  const float* b_proj = (const float*)d_in[3];
  float* out = (float*)d_out;

  bf16* xb     = (bf16*)d_ws;
  bf16* wqkvb  = xb + (size_t)NTOK * CDIM;
  bf16* wprojb = wqkvb + (size_t)CDIM * C3;
  bf16* qkvb   = wprojb + (size_t)CDIM * CDIM;
  bf16* aob    = qkvb + (size_t)NTOK * C3;

  cvt_f32_bf16<<<(NTOK * CDIM / 4) / 256, 256, 0, stream>>>(
      (const float4*)x, (unsigned long long*)xb, NTOK * CDIM / 4);
  cvt_f32_bf16<<<(CDIM * C3 / 4) / 256, 256, 0, stream>>>(
      (const float4*)w_qkv, (unsigned long long*)wqkvb, CDIM * C3 / 4);
  cvt_f32_bf16<<<(CDIM * CDIM / 4) / 256, 256, 0, stream>>>(
      (const float4*)w_proj, (unsigned long long*)wprojb, CDIM * CDIM / 4);

  gemm_wmma<true><<<dim3(C3 / 64, NTOK / 128), 256, 0, stream>>>(
      xb, wqkvb, qkvb, nullptr, NTOK, C3, CDIM);

  rope_kernel<<<(NTOK * NHEAD * 2 * 2 * 16) / 256, 256, 0, stream>>>(qkvb);

  attn_wmma<<<128 * NHEAD * 2, 256, 0, stream>>>(qkvb, aob);

  gemm_wmma<false><<<dim3(CDIM / 64, NTOK / 128), 256, 0, stream>>>(
      aob, wprojb, out, b_proj, NTOK, CDIM, CDIM);

  (void)in_sizes; (void)n_in; (void)out_size; (void)ws_size;
}
